// SigLipAttention_89180700934302
// MI455X (gfx1250) — matrix operand order repaired and checked by reading, not compiled, not run on hardware
//
#include <hip/hip_runtime.h>
#include <hip/hip_bf16.h>

// SigLip attention for MI455X (gfx1250, wave32, WMMA f16 16x16x32).
// B=32, S=576, D=768, H=12, HD=64.

#define BB 32
#define SS 576
#define DD 768
#define HH 12
#define HD 64
#define QSCALE 0.125f   // HD^-0.5

typedef _Float16 half_t;
typedef __attribute__((ext_vector_type(16))) _Float16 v16h;
typedef __attribute__((ext_vector_type(8)))  _Float16 v8h;
typedef __attribute__((ext_vector_type(8)))  float    v8f;

// ---------------------------------------------------------------------------
// Fragment loaders (CDNA5 WMMA 16x16x32 f16 VGPR layouts, CDNA5 ISA)
// A (16x32, MxK): lane L -> row m = L&15; L<16 holds K={kb+0..7, kb+16..23},
//                 L>=16 holds K={kb+8..15, kb+24..31}.
// B (32x16, KxN): source stored as Bsrc[n][k] (N rows, K contiguous).
//                 lane L -> col n = L&15; same K split as A.
// C/D (16x16 f32): lane L -> col n = L&15; vgpr r -> row (L<16 ? r : r+8).
// ---------------------------------------------------------------------------
__device__ __forceinline__ v16h load_a_frag(const half_t* __restrict__ base,
                                            int ld, int kb, int lane) {
  const int m   = lane & 15;
  const int sel = lane >> 4;                       // 0 or 1
  const half_t* p = base + (size_t)m * ld + kb + sel * 8;
  v8h lo = *reinterpret_cast<const v8h*>(p);       // K = kb + sel*8 + 0..7
  v8h hi = *reinterpret_cast<const v8h*>(p + 16);  // K = kb + 16 + sel*8 + 0..7
  v16h a;
#pragma unroll
  for (int i = 0; i < 8; ++i) { a[i] = lo[i]; a[i + 8] = hi[i]; }
  return a;
}

__device__ __forceinline__ v16h load_b_frag(const half_t* __restrict__ base,
                                            int ld, int kb, int lane) {
  const int n   = lane & 15;
  const int sel = lane >> 4;
  const half_t* p = base + (size_t)n * ld + kb + sel * 8;
  v8h lo = *reinterpret_cast<const v8h*>(p);
  v8h hi = *reinterpret_cast<const v8h*>(p + 16);
  v16h b;
#pragma unroll
  for (int i = 0; i < 8; ++i) { b[i] = lo[i]; b[i + 8] = hi[i]; }
  return b;
}

#define WMMA_F16(a, b, c) \
  __builtin_amdgcn_wmma_f32_16x16x32_f16(false, (a), false, (b), (short)0, (c), false, false)

// ---------------------------------------------------------------------------
// Kernel 1: fp32 -> f16 convert (grid-stride)
// ---------------------------------------------------------------------------
__global__ void cvt_f32_to_f16(const float* __restrict__ src,
                               half_t* __restrict__ dst, int n) {
  for (int i = blockIdx.x * blockDim.x + threadIdx.x; i < n;
       i += gridDim.x * blockDim.x)
    dst[i] = (half_t)src[i];
}

// ---------------------------------------------------------------------------
// Kernel 2: fused QKV projection.  Y = X @ W^T + b  (M=B*S, N=D, K=D)
// z=0 -> Q (scaled by QSCALE) into [B,H,S,HD]
// z=1 -> K into [B,H,S,HD]
// z=2 -> V into [B,H,HD,S]   (pre-transposed for P@V B-fragments)
// 256 thr = 8 waves in 4x2; each wave 32x32 output (2x2 reg-blocked WMMAs),
// block tile 128x64.
// ---------------------------------------------------------------------------
__global__ __launch_bounds__(256) void qkv_gemm_kernel(
    const half_t* __restrict__ Xh,
    const half_t* __restrict__ Wq, const half_t* __restrict__ Wk,
    const half_t* __restrict__ Wv,
    const float* __restrict__ bq, const float* __restrict__ bk,
    const float* __restrict__ bv,
    half_t* __restrict__ Qh, half_t* __restrict__ Kh,
    half_t* __restrict__ Vth) {
  const int tid  = threadIdx.x;
  const int wave = tid >> 5;
  const int lane = tid & 31;
  const int m0 = blockIdx.x * 128 + (wave & 3) * 32;  // output row block (32)
  const int n0 = blockIdx.y * 64 + (wave >> 2) * 32;  // output col block (32)
  const int z  = blockIdx.z;

  const half_t* W    = (z == 0) ? Wq : (z == 1) ? Wk : Wv;
  const float*  bias = (z == 0) ? bq : (z == 1) ? bk : bv;

  const half_t* A0 = Xh + (size_t)m0 * DD;
  const half_t* A1 = A0 + (size_t)16 * DD;
  const half_t* B0 = W + (size_t)n0 * DD;
  const half_t* B1 = B0 + (size_t)16 * DD;

  v8f acc00 = {}, acc01 = {}, acc10 = {}, acc11 = {};
#pragma unroll 2
  for (int k = 0; k < DD; k += 32) {
    v16h a0 = load_a_frag(A0, DD, k, lane);
    v16h a1 = load_a_frag(A1, DD, k, lane);
    v16h b0 = load_b_frag(B0, DD, k, lane);
    v16h b1 = load_b_frag(B1, DD, k, lane);
    acc00 = WMMA_F16(a0, b0, acc00);
    acc01 = WMMA_F16(a0, b1, acc01);
    acc10 = WMMA_F16(a1, b0, acc10);
    acc11 = WMMA_F16(a1, b1, acc11);
  }

  const int n  = lane & 15;
  const int rb = (lane < 16) ? 0 : 8;
  const int bat = m0 / SS;            // 32-row tile never crosses batch bound
  const int sb  = m0 - bat * SS;

  v8f accs[2][2] = {{acc00, acc01}, {acc10, acc11}};
#pragma unroll
  for (int j = 0; j < 2; ++j) {
    const int e  = n0 + j * 16 + n;   // feature index
    const int h  = e >> 6;            // head
    const int hd = e & 63;            // within-head dim
    const float bv_ = bias[e];
#pragma unroll
    for (int i = 0; i < 2; ++i) {
#pragma unroll
      for (int r = 0; r < 8; ++r) {
        const int s = sb + i * 16 + rb + r;
        float y = accs[i][j][r] + bv_;
        if (z == 0) {
          y *= QSCALE;
          Qh[(((size_t)bat * HH + h) * SS + s) * HD + hd] = (half_t)y;
        } else if (z == 1) {
          Kh[(((size_t)bat * HH + h) * SS + s) * HD + hd] = (half_t)y;
        } else {
          Vth[(((size_t)bat * HH + h) * HD + hd) * SS + s] = (half_t)y;
        }
      }
    }
  }
}

// ---------------------------------------------------------------------------
// Kernel 3: attention. One block per (b, h, 64-row Q block).
// LDS: scores f32 [64][576] (147456 B) + P f16 [64][576] (73728 B) = 221184 B
// Steps: Q@K^T (WMMA, hoisted Q frags) -> softmax in LDS -> write attn f32
//        -> P@V (WMMA from LDS, 2 col tiles per wave)
// ---------------------------------------------------------------------------
__global__ __launch_bounds__(256) void attn_kernel(
    const half_t* __restrict__ Qh, const half_t* __restrict__ Kh,
    const half_t* __restrict__ Vth,
    float* __restrict__ attn_out, half_t* __restrict__ Ctxh) {
  extern __shared__ char smem[];
  float*  sc = reinterpret_cast<float*>(smem);                 // [64][576]
  half_t* pb = reinterpret_cast<half_t*>(smem + 64 * SS * 4);  // [64][576]
  __shared__ float red[256];
  __shared__ float rowmax[64];
  __shared__ float rowsum[64];

  const int tid  = threadIdx.x;
  const int wave = tid >> 5;
  const int lane = tid & 31;

  const int blk = blockIdx.x;
  const int qb  = blk % (SS / 64);          // 0..8
  const int bh  = blk / (SS / 64);          // 0..B*H-1
  const int q0  = qb * 64;

  const half_t* Qb = Qh  + (size_t)bh * SS * HD;
  const half_t* Kb = Kh  + (size_t)bh * SS * HD;
  const half_t* Vb = Vth + (size_t)bh * HD * SS;

  // Warm L2/WGP$ with the V strip (64x576 f16 = 73728 B) for the P@V phase.
  {
    const char* vpre = reinterpret_cast<const char*>(Vb);
    __builtin_prefetch(vpre + (size_t)tid * 288, 0, 1);
    __builtin_prefetch(vpre + (size_t)tid * 288 + 144, 0, 1);
  }

  // ---- scores = (Q*scale) @ K^T ----
  // wave -> fixed q-row tile qr = wave&3, half of the 36 col tiles (18 each).
  {
    const int qr = wave & 3;
    const int kh = wave >> 2;             // 0 or 1
    const half_t* Arow = Qb + (size_t)(q0 + qr * 16) * HD;
    v16h aq0 = load_a_frag(Arow, HD, 0, lane);   // hoisted Q fragments
    v16h aq1 = load_a_frag(Arow, HD, 32, lane);
    const int n  = lane & 15;
    const int rb = (lane < 16) ? 0 : 8;
    for (int kc = kh * 18; kc < kh * 18 + 18; ++kc) {
      const half_t* Brow = Kb + (size_t)(kc * 16) * HD;
      v16h b0 = load_b_frag(Brow, HD, 0, lane);
      v16h b1 = load_b_frag(Brow, HD, 32, lane);
      v8f acc = {};
      acc = WMMA_F16(aq0, b0, acc);
      acc = WMMA_F16(aq1, b1, acc);
#pragma unroll
      for (int r = 0; r < 8; ++r)
        sc[(qr * 16 + rb + r) * SS + kc * 16 + n] = acc[r];
    }
  }
  __syncthreads();

  // ---- softmax over 576 cols; 4 threads per row, 144 cols each ----
  const int r  = tid & 63;
  const int c0 = (tid >> 6) * 144;
  float pmax = -3.402823466e38f;
  for (int c = 0; c < 144; ++c) pmax = fmaxf(pmax, sc[r * SS + c0 + c]);
  red[tid] = pmax;
  __syncthreads();
  if (tid < 64)
    rowmax[tid] = fmaxf(fmaxf(red[tid], red[tid + 64]),
                        fmaxf(red[tid + 128], red[tid + 192]));
  __syncthreads();
  const float mx = rowmax[r];
  float psum = 0.f;
  for (int c = 0; c < 144; ++c) {
    float e = __expf(sc[r * SS + c0 + c] - mx);
    sc[r * SS + c0 + c] = e;
    psum += e;
  }
  red[tid] = psum;
  __syncthreads();
  if (tid < 64)
    rowsum[tid] = red[tid] + red[tid + 64] + red[tid + 128] + red[tid + 192];
  __syncthreads();
  const float inv = 1.0f / rowsum[r];
  float* arow = attn_out + ((size_t)bh * SS + q0 + r) * SS;
  for (int c = 0; c < 144; ++c) {
    const float p = sc[r * SS + c0 + c] * inv;
    arow[c0 + c]        = p;          // attn output (fp32, d_out)
    pb[r * SS + c0 + c] = (half_t)p;  // f16 P for P@V
  }
  __syncthreads();

  // ---- O = P @ V : wave -> q-row tile qr = wave&3, 2 col tiles (nh=wave>>2) ----
  {
    const int qr = wave & 3;
    const int nh = wave >> 2;             // 0 or 1 -> cols nh*32 .. nh*32+31
    const half_t* Arow = pb + (size_t)(qr * 16) * SS;
    const half_t* B0 = Vb + (size_t)(nh * 32) * SS;
    const half_t* B1 = B0 + (size_t)16 * SS;
    v8f acc0 = {}, acc1 = {};
#pragma unroll 3
    for (int k = 0; k < SS; k += 32) {
      v16h a  = load_a_frag(Arow, SS, k, lane);  // shared by both col tiles
      v16h b0 = load_b_frag(B0, SS, k, lane);
      v16h b1 = load_b_frag(B1, SS, k, lane);
      acc0 = WMMA_F16(a, b0, acc0);
      acc1 = WMMA_F16(a, b1, acc1);
    }
    const int n  = lane & 15;
    const int rb = (lane < 16) ? 0 : 8;
    const int h   = bh % HH;
    const int bat = bh / HH;
    v8f accs[2] = {acc0, acc1};
#pragma unroll
    for (int j = 0; j < 2; ++j) {
#pragma unroll
      for (int rr = 0; rr < 8; ++rr) {
        const int s = q0 + qr * 16 + rb + rr;
        Ctxh[((size_t)bat * SS + s) * DD + h * HD + nh * 32 + j * 16 + n] =
            (half_t)accs[j][rr];
      }
    }
  }
}

// ---------------------------------------------------------------------------
// Kernel 4: output projection.  out = Ctx @ Wo^T + bo  (fp32 result)
// Same 2x2 register blocking as kernel 2.
// ---------------------------------------------------------------------------
__global__ __launch_bounds__(256) void proj_gemm_kernel(
    const half_t* __restrict__ Ctxh, const half_t* __restrict__ Wo,
    const float* __restrict__ bo, float* __restrict__ out) {
  const int tid  = threadIdx.x;
  const int wave = tid >> 5;
  const int lane = tid & 31;
  const int m0 = blockIdx.x * 128 + (wave & 3) * 32;
  const int n0 = blockIdx.y * 64 + (wave >> 2) * 32;

  const half_t* A0 = Ctxh + (size_t)m0 * DD;
  const half_t* A1 = A0 + (size_t)16 * DD;
  const half_t* B0 = Wo + (size_t)n0 * DD;
  const half_t* B1 = B0 + (size_t)16 * DD;

  v8f acc00 = {}, acc01 = {}, acc10 = {}, acc11 = {};
#pragma unroll 2
  for (int k = 0; k < DD; k += 32) {
    v16h a0 = load_a_frag(A0, DD, k, lane);
    v16h a1 = load_a_frag(A1, DD, k, lane);
    v16h b0 = load_b_frag(B0, DD, k, lane);
    v16h b1 = load_b_frag(B1, DD, k, lane);
    acc00 = WMMA_F16(a0, b0, acc00);
    acc01 = WMMA_F16(a0, b1, acc01);
    acc10 = WMMA_F16(a1, b0, acc10);
    acc11 = WMMA_F16(a1, b1, acc11);
  }

  const int n  = lane & 15;
  const int rb = (lane < 16) ? 0 : 8;
  v8f accs[2][2] = {{acc00, acc01}, {acc10, acc11}};
#pragma unroll
  for (int j = 0; j < 2; ++j) {
    const float bv_ = bo[n0 + j * 16 + n];
#pragma unroll
    for (int i = 0; i < 2; ++i) {
#pragma unroll
      for (int r = 0; r < 8; ++r)
        out[(size_t)(m0 + i * 16 + rb + r) * DD + n0 + j * 16 + n] =
            accs[i][j][r] + bv_;
    }
  }
}

// ---------------------------------------------------------------------------
// Host launcher
// ---------------------------------------------------------------------------
extern "C" void kernel_launch(void* const* d_in, const int* in_sizes, int n_in,
                              void* d_out, int out_size, void* d_ws, size_t ws_size,
                              hipStream_t stream) {
  (void)in_sizes; (void)n_in; (void)out_size; (void)ws_size;

  const float* X  = (const float*)d_in[0];
  const float* Wq = (const float*)d_in[1];
  const float* bq = (const float*)d_in[2];
  const float* Wk = (const float*)d_in[3];
  const float* bk = (const float*)d_in[4];
  const float* Wv = (const float*)d_in[5];
  const float* bv = (const float*)d_in[6];
  const float* Wo = (const float*)d_in[7];
  const float* bo = (const float*)d_in[8];

  const size_t NX = (size_t)BB * SS * DD;   // 14,155,776
  const size_t NW = (size_t)DD * DD;        //    589,824

  half_t* w    = (half_t*)d_ws;
  half_t* Xh   = w;               w += NX;
  half_t* Wqh  = w;               w += NW;
  half_t* Wkh  = w;               w += NW;
  half_t* Wvh  = w;               w += NW;
  half_t* Woh  = w;               w += NW;
  half_t* Qh   = w;               w += NX;
  half_t* Kh   = w;               w += NX;
  half_t* Vth  = w;               w += NX;
  half_t* Ctxh = w;               w += NX;

  float* out_p  = (float*)d_out;            // [B,S,D]
  float* attn_p = out_p + NX;               // [B,H,S,S]

  // 1) fp32 -> f16 conversions
  cvt_f32_to_f16<<<(int)((NX + 1023) / 1024), 1024, 0, stream>>>(X,  Xh,  (int)NX);
  cvt_f32_to_f16<<<(int)((NW + 1023) / 1024), 1024, 0, stream>>>(Wq, Wqh, (int)NW);
  cvt_f32_to_f16<<<(int)((NW + 1023) / 1024), 1024, 0, stream>>>(Wk, Wkh, (int)NW);
  cvt_f32_to_f16<<<(int)((NW + 1023) / 1024), 1024, 0, stream>>>(Wv, Wvh, (int)NW);
  cvt_f32_to_f16<<<(int)((NW + 1023) / 1024), 1024, 0, stream>>>(Wo, Woh, (int)NW);

  // 2) QKV projections (z: 0=Q, 1=K, 2=V^T)
  {
    dim3 grid((BB * SS) / 128, DD / 64, 3);
    qkv_gemm_kernel<<<grid, 256, 0, stream>>>(Xh, Wqh, Wkh, Wvh,
                                              bq, bk, bv, Qh, Kh, Vth);
  }

  // 3) attention: one block per (b, h, 64-row q block)
  {
    const int nblk = BB * HH * (SS / 64);             // 3456
    const size_t shmem = (size_t)64 * SS * 4 + (size_t)64 * SS * 2; // 221184 B
    attn_kernel<<<nblk, 256, shmem, stream>>>(Qh, Kh, Vth, attn_p, Ctxh);
  }

  // 4) output projection
  {
    dim3 grid((BB * SS) / 128, DD / 64);
    proj_gemm_kernel<<<grid, 256, 0, stream>>>(Ctxh, Woh, bo, out_p);
  }
}
